// StressXLSTMModel_274877907575
// MI455X (gfx1250) — compile-verified, matrix operand order checked by reading, not run on hardware
//
#include <hip/hip_runtime.h>
#include <hip/hip_bf16.h>

// ---------------------------------------------------------------------------
// xLSTM (sLSTM) encoder forward for MI455X (gfx1250, wave32, WMMA).
// All GEMMs use v_wmma_f32_16x16x32_bf16 (bf16 operands, f32 accumulate).
// ---------------------------------------------------------------------------

#define Bsz 512
#define Tsz 128
#define Dsz 128
#define Hh  4
#define HD  32
#define UP  192
#define BT  (Bsz * Tsz)
#define EPSV 1e-5f

typedef __attribute__((ext_vector_type(16))) __bf16 bf16x16;
typedef __attribute__((ext_vector_type(8)))  float  v8f;

union U32B { uint4 q[2]; bf16x16 v; };

__device__ __forceinline__ unsigned short f2bf(float f) {
    unsigned int u = __float_as_uint(f);
    u += 0x7fffu + ((u >> 16) & 1u);           // round-to-nearest-even
    return (unsigned short)(u >> 16);
}

__device__ __forceinline__ v8f wmma_bf16(bf16x16 a, bf16x16 b, v8f c) {
    return __builtin_amdgcn_wmma_f32_16x16x32_bf16(
        false, a, false, b, (short)0, c, false, false);
}

// A fragment (16x32 bf16): lane l reads row (l&15); halves [hi,hi+8) and
// [16+hi,16+hi+8) with hi = (l>=16)?8:0.  row must be 16B aligned.
__device__ __forceinline__ bf16x16 load_afrag(const unsigned short* row, int lane) {
    const int hi = (lane & 16) ? 8 : 0;
    U32B u;
    u.q[0] = *(const uint4*)(row + hi);
    u.q[1] = *(const uint4*)(row + 16 + hi);
    return u.v;
}

// B fragment: pre-shuffled, 32 contiguous bytes per lane.
__device__ __forceinline__ bf16x16 load_bfrag(const unsigned short* p) {
    U32B u;
    u.q[0] = *(const uint4*)(p);
    u.q[1] = *(const uint4*)(p + 8);
    return u.v;
}

// ---------------------------------------------------------------------------
// Embedding gather
// ---------------------------------------------------------------------------
__global__ void k_embed(const int* __restrict__ tok, const float* __restrict__ emb,
                        float* __restrict__ X) {
    size_t bt = blockIdx.x;
    int d = threadIdx.x;
    int t = tok[bt];
    X[bt * Dsz + d] = emb[(size_t)t * Dsz + d];
}

// ---------------------------------------------------------------------------
// LayerNorm over D=128, one wave per row; optional f32 and bf16 outputs.
// ---------------------------------------------------------------------------
__global__ __launch_bounds__(32) void k_ln(const float* __restrict__ X,
                                           const float* __restrict__ w,
                                           float* __restrict__ outF,
                                           unsigned short* __restrict__ outB) {
    size_t row = blockIdx.x;
    int lane = threadIdx.x;
    const float* xr = X + row * Dsz;
    float v0 = xr[lane], v1 = xr[lane + 32], v2 = xr[lane + 64], v3 = xr[lane + 96];
    float s = v0 + v1 + v2 + v3;
    float ss = v0 * v0 + v1 * v1 + v2 * v2 + v3 * v3;
    for (int o = 16; o; o >>= 1) {
        s  += __shfl_xor(s,  o, 32);
        ss += __shfl_xor(ss, o, 32);
    }
    float mu = s * (1.f / 128.f);
    float var = ss * (1.f / 128.f) - mu * mu;
    float r = rsqrtf(var + EPSV);
    float o0 = (v0 - mu) * r * w[lane];
    float o1 = (v1 - mu) * r * w[lane + 32];
    float o2 = (v2 - mu) * r * w[lane + 64];
    float o3 = (v3 - mu) * r * w[lane + 96];
    if (outF) {
        float* yr = outF + row * Dsz;
        yr[lane] = o0; yr[lane + 32] = o1; yr[lane + 64] = o2; yr[lane + 96] = o3;
    }
    if (outB) {
        unsigned short* yb = outB + row * Dsz;
        yb[lane] = f2bf(o0); yb[lane + 32] = f2bf(o1);
        yb[lane + 64] = f2bf(o2); yb[lane + 96] = f2bf(o3);
    }
}

// ---------------------------------------------------------------------------
// Causal depthwise conv (K=4) + SiLU -> bf16
// ---------------------------------------------------------------------------
__global__ void k_conv(const float* __restrict__ XLN, const float* __restrict__ cw,
                       const float* __restrict__ cb, unsigned short* __restrict__ xcB) {
    size_t bt = blockIdx.x;
    int d = threadIdx.x;
    int t = (int)(bt & (Tsz - 1));
    float a = cb[d];
#pragma unroll
    for (int k = 0; k < 4; ++k) {
        int tt = t + k - 3;
        if (tt >= 0)
            a += XLN[(bt + (size_t)(k - 3)) * Dsz + d] * cw[d * 4 + k];
    }
    a = a / (1.f + __expf(-a));   // silu
    xcB[bt * Dsz + d] = f2bf(a);
}

// ---------------------------------------------------------------------------
// Weight pre-shuffle into WMMA B-fragment layout.
// out[((kt*nTiles+nt)*32+lane)*16 + e], e=2v+p:
//   K = kt*32 + 2v + (v>=4?8:0) + (lane>=16?8:0) + p ;  N = nt*16 + lane%16
// ---------------------------------------------------------------------------
__global__ void k_pack_dense(const float* __restrict__ W, unsigned short* __restrict__ out,
                             int kTiles, int nTiles, int Krows, int Ncols, int rs) {
    int idx = blockIdx.x * blockDim.x + threadIdx.x;
    int total = kTiles * nTiles * 512;
    if (idx >= total) return;
    int e = idx & 15, lane = (idx >> 4) & 31;
    int nt = (idx >> 9) % nTiles, kt = (idx >> 9) / nTiles;
    int v = e >> 1, p = e & 1;
    int K = kt * 32 + 2 * v + (v >= 4 ? 8 : 0) + ((lane & 16) ? 8 : 0) + p;
    int N = nt * 16 + (lane & 15);
    float val = (K < Krows && N < Ncols) ? W[(size_t)K * rs + N] : 0.f;
    out[idx] = f2bf(val);
}

// Block-diagonal pack of two headwise [H][32][32] matrices -> K=128, N=256.
__global__ void k_pack_bdiag2(const float* __restrict__ Wa, const float* __restrict__ Wb,
                              unsigned short* __restrict__ out) {
    int idx = blockIdx.x * blockDim.x + threadIdx.x;     // total 4*16*512 = 32768
    if (idx >= 32768) return;
    int e = idx & 15, lane = (idx >> 4) & 31;
    int nt = (idx >> 9) & 15, kt = (idx >> 9) >> 4;
    int v = e >> 1, p = e & 1;
    int K = kt * 32 + 2 * v + (v >= 4 ? 8 : 0) + ((lane & 16) ? 8 : 0) + p;
    int Ng = nt * 16 + (lane & 15);
    const float* Ws = (Ng < 128) ? Wa : Wb;
    int n = Ng & 127;
    float val = ((K >> 5) == (n >> 5)) ? Ws[(n >> 5) * 1024 + (K & 31) * 32 + (n & 31)] : 0.f;
    out[idx] = f2bf(val);
}

// ---------------------------------------------------------------------------
// Generic bf16 WMMA GEMM: one wave computes a 16x16 f32 tile.
// A: bf16 row-major [M, lda]; Bp: packed fragments; C: f32 [M, ldc].
// grid = (M/16, Nused/16)
// ---------------------------------------------------------------------------
__global__ __launch_bounds__(32) void k_gemm(const unsigned short* __restrict__ A, int lda,
                                             const unsigned short* __restrict__ Bp, int nTilesB,
                                             float* __restrict__ C, int ldc, int kTiles) {
    int lane = threadIdx.x;
    int mt = blockIdx.x, nt = blockIdx.y;
    int m0 = lane & 15, hiM = (lane & 16) ? 8 : 0;
    v8f acc;
#pragma unroll
    for (int j = 0; j < 8; ++j) acc[j] = 0.f;
    const unsigned short* arow = A + (size_t)(mt * 16 + m0) * lda;
    for (int kt = 0; kt < kTiles; ++kt) {
        bf16x16 a = load_afrag(arow + kt * 32, lane);
        bf16x16 b = load_bfrag(Bp + ((size_t)(kt * nTilesB + nt) * 32 + lane) * 16);
        acc = wmma_bf16(a, b, acc);
    }
#pragma unroll
    for (int j = 0; j < 8; ++j) {
        int r = mt * 16 + j + hiM;
        C[(size_t)r * ldc + nt * 16 + m0] = acc[j];
    }
}

// ---------------------------------------------------------------------------
// sLSTM scan. One wave owns 16 batch rows x one head across all T=128 steps.
// Rk B-fragments stay resident in VGPRs (8 x v16bf); hidden state transposed
// through LDS (bf16) each step for the next A fragment.
// gates layout: [BT, 512] = [i_all | f_all | z_all | o_all], each 128 = H*HD.
// ---------------------------------------------------------------------------
__global__ __launch_bounds__(32) void k_scan(const float* __restrict__ gates,
                                             const unsigned short* __restrict__ rkP,
                                             const float* __restrict__ cellb,
                                             float* __restrict__ Y) {
    __shared__ __align__(16) unsigned short hl[16 * 32];   // bf16 h state
    int lane = threadIdx.x;
    int btile = blockIdx.x;                                // 0..31
    int head = blockIdx.y;                                 // 0..3
    for (int i = lane; i < 512; i += 32) hl[i] = 0;
    int m0 = lane & 15, hiM = (lane & 16) ? 8 : 0;

    bf16x16 bf[8];
#pragma unroll
    for (int nt = 0; nt < 8; ++nt)
        bf[nt] = load_bfrag(rkP + ((size_t)(head * 8 + nt) * 32 + lane) * 16);

    float cb[8];
#pragma unroll
    for (int nt = 0; nt < 8; ++nt) {
        int g = nt >> 1, d = ((nt & 1) << 4) + m0;
        cb[nt] = cellb[head * 128 + g * 32 + d];
    }

    float cs[16], ns[16], ms[16];
#pragma unroll
    for (int i = 0; i < 16; ++i) { cs[i] = 0.f; ns[i] = 0.f; ms[i] = 0.f; }

    int b0 = btile * 16;
    for (int t = 0; t < Tsz; ++t) {
        v8f acc[8];
#pragma unroll
        for (int nt = 0; nt < 8; ++nt) {
            int g = nt >> 1;
            int coff = g * 128 + head * 32 + ((nt & 1) << 4) + m0;
#pragma unroll
            for (int j = 0; j < 8; ++j) {
                size_t row = (size_t)(b0 + j + hiM) * Tsz + t;
                acc[nt][j] = gates[row * 512 + coff] + cb[nt];
            }
        }
        // A fragment = previous h (bf16) from LDS
        bf16x16 a;
        {
            const unsigned short* rowp = hl + m0 * 32;
            int hi = (lane & 16) ? 8 : 0;
            U32B u;
            u.q[0] = *(const uint4*)(rowp + hi);
            u.q[1] = *(const uint4*)(rowp + 16 + hi);
            a = u.v;
        }
#pragma unroll
        for (int nt = 0; nt < 8; ++nt) acc[nt] = wmma_bf16(a, bf[nt], acc[nt]);

#pragma unroll
        for (int ct = 0; ct < 2; ++ct) {
#pragma unroll
            for (int j = 0; j < 8; ++j) {
                int si = ct * 8 + j;
                float ir = acc[0 + ct][j];
                float fr = acc[2 + ct][j];
                float zr = acc[4 + ct][j];
                float og = acc[6 + ct][j];
                float mn = fmaxf(fr + ms[si], ir);
                float ig = __expf(ir - mn);
                float fg = __expf(fr + ms[si] - mn);
                float cn = fg * cs[si] + ig * tanhf(zr);
                float nn = fg * ns[si] + ig;
                float hv = (1.f / (1.f + __expf(-og))) * cn / nn;
                cs[si] = cn; ns[si] = nn; ms[si] = mn;
                int d = ct * 16 + m0;
                Y[((size_t)(b0 + j + hiM) * Tsz + t) * Dsz + head * HD + d] = hv;
                hl[(j + hiM) * 32 + d] = f2bf(hv);
            }
        }
        __syncthreads();
    }
}

// ---------------------------------------------------------------------------
// Headwise groupnorm (over HD=32) * gn_w, residual add into X.
// ---------------------------------------------------------------------------
__global__ __launch_bounds__(128) void k_gnres(const float* __restrict__ Y,
                                               const float* __restrict__ gw,
                                               float* __restrict__ X) {
    size_t bt = blockIdx.x;
    int tid = threadIdx.x;
    float v = Y[bt * Dsz + tid];
    float s = v, ss = v * v;
    for (int o = 16; o; o >>= 1) {        // reduce within the 32-lane head group
        s  += __shfl_xor(s,  o, 32);
        ss += __shfl_xor(ss, o, 32);
    }
    float mu = s * (1.f / 32.f);
    float var = ss * (1.f / 32.f) - mu * mu;
    float r = rsqrtf(var + EPSV);
    X[bt * Dsz + tid] += (v - mu) * r * gw[tid];
}

// ---------------------------------------------------------------------------
// FFN: gelu(gate) * up  -> bf16
// ---------------------------------------------------------------------------
__global__ void k_ffn_act(const float* __restrict__ raw, const float* __restrict__ b1,
                          unsigned short* __restrict__ act) {
    size_t idx = (size_t)blockIdx.x * blockDim.x + threadIdx.x;
    if (idx >= (size_t)BT * UP) return;
    size_t bt = idx / UP;
    int u = (int)(idx % UP);
    float g  = raw[bt * (2 * UP) + u] + b1[u];
    float up = raw[bt * (2 * UP) + UP + u] + b1[UP + u];
    float gl = 0.5f * g * (1.f + tanhf(0.7978845608f * (g + 0.044715f * g * g * g)));
    act[idx] = f2bf(gl * up);
}

__global__ void k_resid2(float* __restrict__ X, const float* __restrict__ O,
                         const float* __restrict__ b2) {
    size_t idx = (size_t)blockIdx.x * blockDim.x + threadIdx.x;
    if (idx >= (size_t)BT * Dsz) return;
    X[idx] += O[idx] + b2[idx & (Dsz - 1)];
}

// ---------------------------------------------------------------------------
// Mean pool over T, then 2-layer classifier head.
// ---------------------------------------------------------------------------
__global__ __launch_bounds__(128) void k_pool(const float* __restrict__ XLN,
                                              float* __restrict__ pooled) {
    int b = blockIdx.x, d = threadIdx.x;
    float s = 0.f;
    for (int t = 0; t < Tsz; ++t)
        s += XLN[((size_t)b * Tsz + t) * Dsz + d];
    pooled[(size_t)b * Dsz + d] = s * (1.f / (float)Tsz);
}

__global__ __launch_bounds__(64) void k_cls(const float* __restrict__ pooled,
                                            const float* __restrict__ w1, const float* __restrict__ b1,
                                            const float* __restrict__ w2, const float* __restrict__ b2,
                                            float* __restrict__ out) {
    __shared__ float h1[64];
    int b = blockIdx.x, j = threadIdx.x;
    float s = b1[j];
    for (int d = 0; d < Dsz; ++d)
        s += pooled[(size_t)b * Dsz + d] * w1[d * 64 + j];
    h1[j] = fmaxf(s, 0.f);
    __syncthreads();
    if (j < 2) {
        float o = b2[j];
        for (int k = 0; k < 64; ++k) o += h1[k] * w2[k * 2 + j];
        out[(size_t)b * 2 + j] = o;
    }
}

// ---------------------------------------------------------------------------
// Host-side orchestration
// ---------------------------------------------------------------------------
extern "C" void kernel_launch(void* const* d_in, const int* in_sizes, int n_in,
                              void* d_out, int out_size, void* d_ws, size_t ws_size,
                              hipStream_t stream) {
    const int*   tokens = (const int*)  d_in[0];
    const float* emb    = (const float*)d_in[1];
    const float* ln1_w  = (const float*)d_in[2];
    const float* conv_w = (const float*)d_in[3];
    const float* conv_b = (const float*)d_in[4];
    const float* Wi     = (const float*)d_in[5];
    const float* Wf     = (const float*)d_in[6];
    const float* Wz     = (const float*)d_in[7];
    const float* Wo     = (const float*)d_in[8];
    const float* Rk     = (const float*)d_in[9];
    const float* cell_b = (const float*)d_in[10];
    const float* gn_w   = (const float*)d_in[11];
    const float* ln2_w  = (const float*)d_in[12];
    const float* ff_w1  = (const float*)d_in[13];
    const float* ff_b1  = (const float*)d_in[14];
    const float* ff_w2  = (const float*)d_in[15];
    const float* ff_b2  = (const float*)d_in[16];
    const float* pn_w   = (const float*)d_in[17];
    const float* cw1    = (const float*)d_in[18];
    const float* cb1    = (const float*)d_in[19];
    const float* cw2    = (const float*)d_in[20];
    const float* cb2    = (const float*)d_in[21];
    float* out = (float*)d_out;

    // ---- workspace carve-up (256B aligned) ----
    char* w = (char*)d_ws;
    auto carve = [&](size_t bytes) {
        char* p = w;
        w += (bytes + 255) & ~(size_t)255;
        return p;
    };
    float*          X     = (float*)         carve((size_t)BT * Dsz * 4);
    float*          XLN   = (float*)         carve((size_t)BT * Dsz * 4);
    unsigned short* XLNB  = (unsigned short*)carve((size_t)BT * Dsz * 2);
    unsigned short* XCB   = (unsigned short*)carve((size_t)BT * Dsz * 2);
    float*          GATES = (float*)         carve((size_t)BT * 512 * 4);   // aliased below
    float*          YBUF  = (float*)         carve((size_t)BT * Dsz * 4);   // aliased below
    float*          POOLED= (float*)         carve((size_t)Bsz * Dsz * 4);
    unsigned short* pIF   = (unsigned short*)carve(32768 * 2);
    unsigned short* pZO   = (unsigned short*)carve(32768 * 2);
    unsigned short* pRK   = (unsigned short*)carve(16384 * 2);
    unsigned short* pW1   = (unsigned short*)carve(49152 * 2);
    unsigned short* pW2   = (unsigned short*)carve(24576 * 2);
    // aliases (lifetimes disjoint): FFN temporaries reuse GATES; act reuses YBUF
    float*          FFNRAW = GATES;                               // [BT,384]
    float*          FFNOUT = GATES + (size_t)BT * (2 * UP);       // [BT,128]
    unsigned short* ACTB   = (unsigned short*)YBUF;               // [BT,192] bf16

    // ---- embedding ----
    k_embed<<<BT, Dsz, 0, stream>>>(tokens, emb, X);

    for (int b = 0; b < 2; ++b) {
        const float* Wi_b = Wi + (size_t)b * Hh * HD * HD;
        const float* Wf_b = Wf + (size_t)b * Hh * HD * HD;
        const float* Wz_b = Wz + (size_t)b * Hh * HD * HD;
        const float* Wo_b = Wo + (size_t)b * Hh * HD * HD;

        // LN1 (f32 for conv, bf16 for z/o GEMM A)
        k_ln<<<BT, 32, 0, stream>>>(X, ln1_w + b * Dsz, XLN, XLNB);
        // causal depthwise conv + silu -> bf16 (A for i/f GEMM)
        k_conv<<<BT, Dsz, 0, stream>>>(XLN, conv_w + (size_t)b * Dsz * 4,
                                       conv_b + b * Dsz, XCB);
        // pack gate weights (block-diagonal, [i|f] and [z|o]) and Rk per head
        k_pack_bdiag2<<<128, 256, 0, stream>>>(Wi_b, Wf_b, pIF);
        k_pack_bdiag2<<<128, 256, 0, stream>>>(Wz_b, Wo_b, pZO);
        for (int h = 0; h < Hh; ++h)
            k_pack_dense<<<16, 256, 0, stream>>>(Rk + ((size_t)b * Hh + h) * HD * 128,
                                                 pRK + h * 4096, 1, 8, 32, 128, 128);
        // gate GEMMs: [BT,128] x [128,256] -> gates[:, 0:256] and [:, 256:512]
        k_gemm<<<dim3(BT / 16, 16), 32, 0, stream>>>(XCB, Dsz, pIF, 16, GATES, 512, 4);
        k_gemm<<<dim3(BT / 16, 16), 32, 0, stream>>>(XLNB, Dsz, pZO, 16, GATES + 256, 512, 4);
        // recurrent scan (WMMA in the T-loop)
        k_scan<<<dim3(Bsz / 16, Hh), 32, 0, stream>>>(GATES, pRK,
                                                      cell_b + (size_t)b * Hh * 128, YBUF);
        // headwise groupnorm + residual
        k_gnres<<<BT, Dsz, 0, stream>>>(YBUF, gn_w + b * Dsz, X);

        // FFN
        k_ln<<<BT, 32, 0, stream>>>(X, ln2_w + b * Dsz, nullptr, XLNB);
        k_pack_dense<<<192, 256, 0, stream>>>(ff_w1 + (size_t)b * Dsz * 2 * UP, pW1,
                                              4, 24, 128, 2 * UP, 2 * UP);
        k_gemm<<<dim3(BT / 16, 24), 32, 0, stream>>>(XLNB, Dsz, pW1, 24, FFNRAW, 2 * UP, 4);
        k_ffn_act<<<(int)(((size_t)BT * UP + 255) / 256), 256, 0, stream>>>(
            FFNRAW, ff_b1 + b * 2 * UP, ACTB);
        k_pack_dense<<<96, 256, 0, stream>>>(ff_w2 + (size_t)b * UP * Dsz, pW2,
                                             6, 8, UP, Dsz, Dsz);
        k_gemm<<<dim3(BT / 16, 8), 32, 0, stream>>>(ACTB, UP, pW2, 8, FFNOUT, Dsz, 6);
        k_resid2<<<(int)(((size_t)BT * Dsz + 255) / 256), 256, 0, stream>>>(
            X, FFNOUT, ff_b2 + b * Dsz);
    }

    // post-norm, pool, classifier
    k_ln<<<BT, 32, 0, stream>>>(X, pn_w, XLN, nullptr);
    k_pool<<<Bsz, Dsz, 0, stream>>>(XLN, POOLED);
    k_cls<<<Bsz, 64, 0, stream>>>(POOLED, cw1, cb1, cw2, cb2, out);
    (void)in_sizes; (void)n_in; (void)out_size; (void)ws_size;
}
